// Attention_51127290692370
// MI455X (gfx1250) — matrix operand order repaired and checked by reading, not compiled, not run on hardware
//
#include <hip/hip_runtime.h>
#include <hip/hip_bf16.h>

#define DIM   1024
#define SEQ   2048
#define BATCH 2
#define HEADS 16
#define HD    64
#define NTOK  (BATCH*SEQ)   // 4096

typedef __bf16 bf16;
typedef __attribute__((ext_vector_type(16))) __bf16 bf16x16;
typedef __attribute__((ext_vector_type(8)))  float  f32x8;

#define F32X8_ZERO {0.f,0.f,0.f,0.f,0.f,0.f,0.f,0.f}

// ---------------------------------------------------------------------------
// WMMA fragment helpers (wave32, contiguous k order k = 16*(lane/16) + e)
// ---------------------------------------------------------------------------

__device__ __forceinline__ f32x8 wmma_bf16(bf16x16 a, bf16x16 b, f32x8 c) {
  return __builtin_amdgcn_wmma_f32_16x16x32_bf16(false, a, false, b, (short)0,
                                                 c, false, false);
}

// A fragment: 16x32 bf16 tile, src row-major with leading dim ld.
// lane L: row = L%16; element e: K = 16*(L/16) + e
__device__ __forceinline__ bf16x16 frag_a(const bf16* p0, int ld, int lane) {
  const int ln = lane & 15, half = lane >> 4;
  const bf16* p = p0 + ln * ld + half * 16;
  union { uint4 u[2]; bf16x16 v; } f;
  f.u[0] = *(const uint4*)(p);
  f.u[1] = *(const uint4*)(p + 8);
  return f.v;
}

// B fragment with TRANSPOSED source: logical B[k][n] = src[n*ld + k]
__device__ __forceinline__ bf16x16 frag_b_t(const bf16* p0, int ld, int lane) {
  const int ln = lane & 15, half = lane >> 4;
  const bf16* p = p0 + ln * ld + half * 16;
  union { uint4 u[2]; bf16x16 v; } f;
  f.u[0] = *(const uint4*)(p);
  f.u[1] = *(const uint4*)(p + 8);
  return f.v;
}

// B fragment, row-major source src[k*ld + n] (strided gather).
__device__ __forceinline__ bf16x16 frag_b(const bf16* p0, int ld, int lane) {
  const int ln = lane & 15, kb = (lane >> 4) * 16;
  bf16x16 f;
#pragma unroll
  for (int e = 0; e < 16; ++e) f[e] = p0[(kb + e) * ld + ln];
  return f;
}

// ---------------------------------------------------------------------------
// CDNA5 async memory->LDS copy (ASYNCcnt-tracked; bypasses VGPRs).
// LDS offset = low 32 bits of generic pointer (aperture rule, ISA 10.2).
// ---------------------------------------------------------------------------
__device__ __forceinline__ void async_ld128(unsigned ldsOff, const void* gaddr) {
  asm volatile("global_load_async_to_lds_b128 %0, %1, off"
               :: "v"(ldsOff), "v"(gaddr)
               : "memory");
}
__device__ __forceinline__ void wait_async0() {
  asm volatile("s_wait_asynccnt 0x0" ::: "memory");
}

// ---------------------------------------------------------------------------
// Elementwise convert f32 -> bf16
// ---------------------------------------------------------------------------
__global__ void cvt_f32_bf16(const float* __restrict__ src,
                             bf16* __restrict__ dst, int n) {
  int i = blockIdx.x * blockDim.x + threadIdx.x;
  if (i < n) dst[i] = (bf16)src[i];
}

// ---------------------------------------------------------------------------
// Repack DxD f32 weight into WMMA-B-fragment-native bf16 layout:
// tile (kb, nt): offset = ((kb*64+nt)*512 + lane*16 + e),
//   k = kb*32 + 16*(lane/16) + e,  n = nt*16 + lane%16
// -> one contiguous 32-byte load per lane per WMMA B operand.
// ---------------------------------------------------------------------------
__global__ void pack_w(const float* __restrict__ W, bf16* __restrict__ WP) {
  int o      = blockIdx.x * 256 + threadIdx.x;
  int tile   = o >> 9;
  int within = o & 511;
  int lane   = within >> 4;
  int e      = within & 15;
  int kb = tile >> 6;
  int nt = tile & 63;
  int k  = kb * 32 + ((lane >> 4) << 4) + e;
  int n  = nt * 16 + (lane & 15);
  WP[o] = (bf16)W[k * DIM + n];
}

// ---------------------------------------------------------------------------
// Row-panel GEMM: C[16 x 1024] = A[16 x 1024] * W[1024 x 1024], software-
// pipelined by one K-step so WMMAs never wait on their own operand loads.
// Optional fused full-row LayerNorm epilogue; bf16 or f32 output.
// ---------------------------------------------------------------------------
__global__ void __launch_bounds__(256)
gemm_proj(const bf16* __restrict__ A, const bf16* __restrict__ WP,
          const float* __restrict__ gamma, const float* __restrict__ beta,
          bf16* __restrict__ outB, float* __restrict__ outF, int do_ln) {
  __shared__ __align__(16) bf16 sA[16 * DIM];     // 32 KB A panel
  __shared__ float redS[8][16];
  __shared__ float redQ[8][16];
  __shared__ float sMu[16], sRs[16];

  const int tid  = threadIdx.x;
  const int w    = tid >> 5, lane = tid & 31;
  const int half = lane >> 4, ln = lane & 15;
  const int tokBase = blockIdx.x * 16;

  // Async-stage the 16x1024 A panel straight into LDS (ASYNCcnt path).
  {
    const bf16* g = A + (size_t)tokBase * DIM;
    const unsigned sBase = (unsigned)(size_t)(void*)sA;
    for (int i = tid; i < 16 * DIM / 8; i += 256)
      async_ld128(sBase + (unsigned)i * 16u, g + i * 8);
  }
  wait_async0();
  __syncthreads();

  f32x8 acc[8];
#pragma unroll
  for (int t = 0; t < 8; ++t) acc[t] = (f32x8)F32X8_ZERO;

  // Prologue: fragments for kb = 0.
  bf16x16 a = frag_a(sA, DIM, lane);
  bf16x16 b[8];
#pragma unroll
  for (int t = 0; t < 8; ++t)
    b[t] = *(const bf16x16*)(WP + ((size_t)(w * 8 + t) << 9) + lane * 16);

  for (int kb = 0; kb < DIM / 32; ++kb) {
    const int kn = (kb + 1 < DIM / 32) ? kb + 1 : kb;   // clamp (dead at end)
    // Issue next iteration's operand loads first...
    bf16x16 a_n = frag_a(sA + kn * 32, DIM, lane);
    bf16x16 b_n[8];
#pragma unroll
    for (int t = 0; t < 8; ++t)
      b_n[t] = *(const bf16x16*)(WP + ((size_t)(kn * 64 + w * 8 + t) << 9) +
                                 lane * 16);
    if (kb + 2 < DIM / 32)   // warm L2->L0 two steps ahead
      __builtin_prefetch(WP + ((size_t)((kb + 2) * 64 + w * 8) << 9), 0, 1);
    // ...then compute on the already-resident fragments.
#pragma unroll
    for (int t = 0; t < 8; ++t)
      acc[t] = wmma_bf16(a, b[t], acc[t]);
    a = a_n;
#pragma unroll
    for (int t = 0; t < 8; ++t) b[t] = b_n[t];
  }

  if (do_ln) {
    // Per-row sum/sumsq: lane partials -> xor-shuffle over 16-lane row group.
#pragma unroll
    for (int v = 0; v < 8; ++v) {
      float s = 0.f, q = 0.f;
#pragma unroll
      for (int t = 0; t < 8; ++t) { float x = acc[t][v]; s += x; q += x * x; }
#pragma unroll
      for (int m = 1; m < 16; m <<= 1) {
        s += __shfl_xor(s, m, 32);
        q += __shfl_xor(q, m, 32);
      }
      if (ln == 0) {
        const int r = v + half * 8;
        redS[w][r] = s;
        redQ[w][r] = q;
      }
    }
    __syncthreads();
    if (tid < 16) {
      float s = 0.f, q = 0.f;
      for (int ww = 0; ww < 8; ++ww) { s += redS[ww][tid]; q += redQ[ww][tid]; }
      float mu  = s * (1.0f / DIM);
      float var = q * (1.0f / DIM) - mu * mu;
      sMu[tid] = mu;
      sRs[tid] = rsqrtf(var + 1e-5f);
    }
    __syncthreads();
  }

  // Epilogue (C/D layout: row = v + 8*half, col = lane%16).
#pragma unroll
  for (int t = 0; t < 8; ++t) {
    const int colBase = w * 128 + t * 16 + ln;
#pragma unroll
    for (int v = 0; v < 8; ++v) {
      const int r = v + half * 8;
      float val = acc[t][v];
      if (do_ln) val = (val - sMu[r]) * sRs[r] * gamma[colBase] + beta[colBase];
      const size_t o = (size_t)(tokBase + r) * DIM + colBase;
      if (outF) outF[o] = val;
      else      outB[o] = (bf16)val;
    }
  }
}

// ---------------------------------------------------------------------------
// Flash attention: one block per (batch, head, 64-query rows); 8 waves.
// K/V double-buffered in LDS via async copies; Q lives in registers.
// ---------------------------------------------------------------------------
__device__ __forceinline__ void stage_kv(const bf16* __restrict__ K,
                                         const bf16* __restrict__ V,
                                         bf16* dK, bf16* dV,
                                         size_t tokRow, size_t headCol,
                                         int tid) {
  const unsigned kBase = (unsigned)(size_t)(void*)dK;
  const unsigned vBase = (unsigned)(size_t)(void*)dV;
#pragma unroll
  for (int i = tid; i < 512; i += 256) {
    const int r = i >> 3, c = i & 7;
    const size_t g = (tokRow + r) * DIM + headCol + c * 8;
    async_ld128(kBase + (unsigned)i * 16u, K + g);
    async_ld128(vBase + (unsigned)i * 16u, V + g);
  }
}

__global__ void __launch_bounds__(256)
flash_attn(const bf16* __restrict__ Q, const bf16* __restrict__ K,
           const bf16* __restrict__ V, bf16* __restrict__ O) {
  __shared__ __align__(16) bf16 sK[2][64 * HD];   // 2 x 8 KB
  __shared__ __align__(16) bf16 sV[2][64 * HD];   // 2 x 8 KB
  __shared__ __align__(16) float sS[64 * 64];     // 16 KB scores
  __shared__ __align__(16) bf16 sP[64 * 64];      // 8 KB probabilities
  __shared__ float sM[64], sL[64], sAlpha[64], sPart[4][64];

  const int tid  = threadIdx.x;
  const int w    = tid >> 5, lane = tid & 31;
  const int half = lane >> 4, ln = lane & 15;
  const int mt = w >> 1;            // M tile 0..3
  const int n0 = (w & 1) * 2;       // first of two N tiles
  const int h  = blockIdx.y, b = blockIdx.z;
  const size_t qtok    = (size_t)b * SEQ + blockIdx.x * 64;
  const size_t headCol = (size_t)h * HD;

  // Q fragments straight from global into registers: each wave only ever
  // multiplies its own 16 query rows (2 K-chunks of hd=64).
  bf16x16 qf[2];
#pragma unroll
  for (int kc = 0; kc < 2; ++kc)
    qf[kc] = frag_a(Q + (qtok + mt * 16) * DIM + headCol + kc * 32, DIM, lane);

  if (tid < 64) { sM[tid] = -1e30f; sL[tid] = 0.f; }

  f32x8 acc[2];
  acc[0] = (f32x8)F32X8_ZERO;
  acc[1] = (f32x8)F32X8_ZERO;

  // Prologue: async-stage key block 0 into buffer 0.
  stage_kv(K, V, &sK[0][0], &sV[0][0], (size_t)b * SEQ, headCol, tid);

  for (int jb = 0; jb < SEQ / 64; ++jb) {
    const int cur = jb & 1;
    wait_async0();          // this wave's staged K/V landed in LDS
    __syncthreads();        // everyone's staged K/V visible; prev buf free
    if (jb + 1 < SEQ / 64)  // overlap next block's staging with compute
      stage_kv(K, V, &sK[cur ^ 1][0], &sV[cur ^ 1][0],
               (size_t)b * SEQ + (size_t)(jb + 1) * 64, headCol, tid);

    // S = Q * K^T (scale is exactly 1.0 per reference).
#pragma unroll
    for (int nt = 0; nt < 2; ++nt) {
      const int n = n0 + nt;
      f32x8 s = (f32x8)F32X8_ZERO;
#pragma unroll
      for (int kc = 0; kc < 2; ++kc) {
        bf16x16 bb = frag_b_t(&sK[cur][0] + n * 16 * HD + kc * 32, HD, lane);
        s = wmma_bf16(qf[kc], bb, s);
      }
#pragma unroll
      for (int v = 0; v < 8; ++v)
        sS[(mt * 16 + v + half * 8) * 64 + n * 16 + ln] = s[v];
    }
    __syncthreads();

    // Online softmax. Phase A: per-quarter row max.
    {
      const int r = tid & 63, qt = tid >> 6;
      float pm = -1e30f;
      for (int c = qt * 16; c < qt * 16 + 16; ++c)
        pm = fmaxf(pm, sS[r * 64 + c]);
      sPart[qt][r] = pm;
    }
    __syncthreads();
    // Phase B: new running max + rescale factor.
    if (tid < 64) {
      float mj = fmaxf(fmaxf(sPart[0][tid], sPart[1][tid]),
                       fmaxf(sPart[2][tid], sPart[3][tid]));
      float mnew = fmaxf(sM[tid], mj);
      sAlpha[tid] = __expf(sM[tid] - mnew);
      sM[tid] = mnew;
    }
    __syncthreads();
    // Phase C: P = exp(S - m) in bf16, partial row sums.
    {
      const int r = tid & 63, qt = tid >> 6;
      const float mrow = sM[r];
      float ps = 0.f;
      for (int c = qt * 16; c < qt * 16 + 16; ++c) {
        float p = __expf(sS[r * 64 + c] - mrow);
        sP[r * 64 + c] = (bf16)p;
        ps += p;
      }
      sPart[qt][r] = ps;
    }
    __syncthreads();
    // Phase D: update running denominator.
    if (tid < 64)
      sL[tid] = sL[tid] * sAlpha[tid] +
                sPart[0][tid] + sPart[1][tid] + sPart[2][tid] + sPart[3][tid];

    // Rescale O accumulators by alpha, then O += P * V.
#pragma unroll
    for (int nt = 0; nt < 2; ++nt)
#pragma unroll
      for (int v = 0; v < 8; ++v)
        acc[nt][v] *= sAlpha[mt * 16 + v + half * 8];
#pragma unroll
    for (int kc = 0; kc < 2; ++kc) {
      bf16x16 a = frag_a(sP + mt * 16 * 64 + kc * 32, 64, lane);
#pragma unroll
      for (int nt = 0; nt < 2; ++nt) {
        bf16x16 bb = frag_b(&sV[cur][0] + kc * 32 * HD + (n0 + nt) * 16, HD,
                            lane);
        acc[nt] = wmma_bf16(a, bb, acc[nt]);
      }
    }
    // Next iteration's wait+barrier fences buffer reuse.
  }

  __syncthreads();   // make final sL visible to all threads

  // Epilogue: O /= l, store bf16 into [tok][h*64 + d].
#pragma unroll
  for (int nt = 0; nt < 2; ++nt)
#pragma unroll
    for (int v = 0; v < 8; ++v) {
      const int r = mt * 16 + v + half * 8;
      const float val = acc[nt][v] / sL[r];
      O[(qtok + r) * DIM + headCol + (n0 + nt) * 16 + ln] = (bf16)val;
    }
}

// ---------------------------------------------------------------------------
extern "C" void kernel_launch(void* const* d_in, const int* in_sizes, int n_in,
                              void* d_out, int out_size, void* d_ws,
                              size_t ws_size, hipStream_t stream) {
  (void)in_sizes; (void)n_in; (void)out_size; (void)ws_size;
  const float* x    = (const float*)d_in[0];
  const float* Wq   = (const float*)d_in[1];
  const float* Wk   = (const float*)d_in[2];
  const float* Wv   = (const float*)d_in[3];
  const float* Wo   = (const float*)d_in[4];
  const float* qgam = (const float*)d_in[5];
  const float* qbet = (const float*)d_in[6];
  const float* kgam = (const float*)d_in[7];
  const float* kbet = (const float*)d_in[8];

  char* ws = (char*)d_ws;
  const size_t MB = 1024 * 1024;
  bf16* xb  = (bf16*)(ws + 0);        // 8 MB
  bf16* WqP = (bf16*)(ws + 8  * MB);  // 2 MB
  bf16* WkP = (bf16*)(ws + 10 * MB);  // 2 MB
  bf16* WvP = (bf16*)(ws + 12 * MB);  // 2 MB
  bf16* WoP = (bf16*)(ws + 14 * MB);  // 2 MB
  bf16* qB  = (bf16*)(ws + 16 * MB);  // 8 MB
  bf16* kB  = (bf16*)(ws + 24 * MB);  // 8 MB
  bf16* vB  = (bf16*)(ws + 32 * MB);  // 8 MB
  bf16* aO  = (bf16*)(ws + 40 * MB);  // 8 MB

  cvt_f32_bf16<<<NTOK * DIM / 256, 256, 0, stream>>>(x, xb, NTOK * DIM);
  pack_w<<<DIM * DIM / 256, 256, 0, stream>>>(Wq, WqP);
  pack_w<<<DIM * DIM / 256, 256, 0, stream>>>(Wk, WkP);
  pack_w<<<DIM * DIM / 256, 256, 0, stream>>>(Wv, WvP);
  pack_w<<<DIM * DIM / 256, 256, 0, stream>>>(Wo, WoP);

  gemm_proj<<<NTOK / 16, 256, 0, stream>>>(xb, WqP, qgam, qbet, qB, nullptr, 1);
  gemm_proj<<<NTOK / 16, 256, 0, stream>>>(xb, WkP, kgam, kbet, kB, nullptr, 1);
  gemm_proj<<<NTOK / 16, 256, 0, stream>>>(xb, WvP, nullptr, nullptr, vB,
                                           nullptr, 0);

  flash_attn<<<dim3(SEQ / 64, HEADS, BATCH), 256, 0, stream>>>(qB, kB, vB, aO);

  gemm_proj<<<NTOK / 16, 256, 0, stream>>>(aO, WoP, nullptr, nullptr, nullptr,
                                           (float*)d_out, 0);
}
